// GNNLayer_85959475462178
// MI455X (gfx1250) — matrix operand order repaired and checked by reading, not compiled, not run on hardware
//
#include <hip/hip_runtime.h>
#include <hip/hip_bf16.h>

typedef __attribute__((ext_vector_type(16))) __bf16 v16bf;
typedef __attribute__((ext_vector_type(8)))  float  v8f;

#define HDIM 128
#define EPSBN 1e-5f
#define SLOPE 0.01f

// ---------------------------------------------------------------------------
// Fragment loaders for V_WMMA_F32_16X16X32_BF16 (wave32; A/B use contiguous k = 16*hi + i, not the ISA table order)
// A: 16x32 (MxK) row-major fp32 source, converted to bf16 in regs.
//    lane L(0..15): row M=L, K = k0 + 0..15; lane L+16: K = k0 + 16..31
// B: 32x16 (KxN), W stored row-major [K,N].
//    lanes 0..15: N=lane, K=k0+0..15 ; lanes 16..31: N=lane-16, K=k0+16..31
// C/D: vgpr j -> row = hi*8 + j, col = lane&15
// ---------------------------------------------------------------------------
__device__ inline v16bf load_frag_a_row(const float* rowp, int hi) {
  const float* p = rowp + hi * 16;
  v16bf a;
#pragma unroll
  for (int i = 0; i < 8; ++i) a[i] = (__bf16)p[i];
#pragma unroll
  for (int i = 0; i < 8; ++i) a[8 + i] = (__bf16)p[8 + i];
  return a;
}

__device__ inline v16bf load_frag_a(const float* A, int lda, int row_base, int k0, int max_row) {
  int lane = threadIdx.x & 31;
  int r = row_base + (lane & 15);
  if (r > max_row) r = max_row;
  return load_frag_a_row(A + (size_t)r * lda + k0, lane >> 4);
}

__device__ inline v16bf load_frag_b(const float* B, int ldb, int k0, int col_base) {
  int lane = threadIdx.x & 31;
  int n = col_base + (lane & 15);
  int hi = lane >> 4;
  const float* p = B + (size_t)(k0 + hi * 16) * ldb + n;
  v16bf b;
#pragma unroll
  for (int i = 0; i < 16; ++i) b[i] = (__bf16)p[(size_t)i * ldb];
  return b;
}

// ---------------------------------------------------------------------------
// 1) scatter-add of source features along edges + per-destination edge count.
//    One wave per edge; each lane handles 4 consecutive floats (float4 load).
// ---------------------------------------------------------------------------
__global__ void scatter_add_kernel(const float* __restrict__ src_feat,
                                   const int* __restrict__ ei,   // [2,E]: [0..E)=src, [E..2E)=dst
                                   float* __restrict__ sums,
                                   float* __restrict__ cnt,
                                   long long E) {
  long long gid = (long long)blockIdx.x * blockDim.x + threadIdx.x;
  long long e = gid >> 5;
  int lane = (int)(gid & 31);
  if (e >= E) return;
  int s = ei[e];
  int d = ei[E + e];
  const float* row = src_feat + (size_t)s * HDIM;
  float* orow = sums + (size_t)d * HDIM;
  int c = lane * 4;
  float4 v = *(const float4*)(row + c);
  atomicAdd(orow + c + 0, v.x);
  atomicAdd(orow + c + 1, v.y);
  atomicAdd(orow + c + 2, v.z);
  atomicAdd(orow + c + 3, v.w);
  if (lane == 0) atomicAdd(cnt + d, 1.0f);
}

__global__ void mean_div_kernel(float* __restrict__ sums, const float* __restrict__ cnt,
                                long long total) {
  long long i = (long long)blockIdx.x * blockDim.x + threadIdx.x;
  if (i >= total) return;
  float c = cnt[i >> 7];  // /128
  sums[i] = sums[i] / fmaxf(c, 1.0f);
}

// ---------------------------------------------------------------------------
// 2) Fused SAGE: out = leaky_relu(mean @ Wl + h @ Wr + bl), M x 128, K = 128.
//    Block = 256 threads = 8 waves; block handles 16 rows, wave w -> cols [16w,16w+16)
// ---------------------------------------------------------------------------
__global__ void sage_gemm_lrelu_kernel(const float* __restrict__ mean,
                                       const float* __restrict__ h,
                                       const float* __restrict__ Wl,
                                       const float* __restrict__ Wr,
                                       const float* __restrict__ bl,
                                       float* __restrict__ out, int M) {
  int wave = threadIdx.x >> 5;
  int row_base = blockIdx.x * 16;
  int col_base = wave * 16;
  v8f acc = {};
#pragma unroll
  for (int k0 = 0; k0 < HDIM; k0 += 32) {
    v16bf a = load_frag_a(mean, HDIM, row_base, k0, M - 1);
    v16bf b = load_frag_b(Wl, HDIM, k0, col_base);
    acc = __builtin_amdgcn_wmma_f32_16x16x32_bf16(false, a, false, b, (short)0, acc, false, false);
  }
#pragma unroll
  for (int k0 = 0; k0 < HDIM; k0 += 32) {
    v16bf a = load_frag_a(h, HDIM, row_base, k0, M - 1);
    v16bf b = load_frag_b(Wr, HDIM, k0, col_base);
    acc = __builtin_amdgcn_wmma_f32_16x16x32_bf16(false, a, false, b, (short)0, acc, false, false);
  }
  int lane = threadIdx.x & 31;
  int n = col_base + (lane & 15);
  int hi = lane >> 4;
  float bias = bl[n];
#pragma unroll
  for (int j = 0; j < 8; ++j) {
    int r = row_base + hi * 8 + j;
    if (r < M) {
      float v = acc[j] + bias;
      out[(size_t)r * HDIM + n] = (v >= 0.0f) ? v : SLOPE * v;
    }
  }
}

// ---------------------------------------------------------------------------
// 3) BatchNorm: column stats (sum, sumsq) then in-place normalize.
// ---------------------------------------------------------------------------
#define BN_ROWS 128
__global__ void bn_stats_kernel(const float* __restrict__ x, float* __restrict__ stats, int M) {
  int c = threadIdx.x;                 // 128 threads, one column each
  int r0 = blockIdx.x * BN_ROWS;
  float s = 0.0f, ss = 0.0f;
  for (int i = 0; i < BN_ROWS; ++i) {
    int r = r0 + i;
    if (r < M) {
      float v = x[(size_t)r * HDIM + c];
      s += v; ss += v * v;
    }
  }
  atomicAdd(&stats[c], s);
  atomicAdd(&stats[HDIM + c], ss);
}

__global__ void bn_norm_kernel(float* __restrict__ x, const float* __restrict__ stats,
                               const float* __restrict__ g, const float* __restrict__ b, int M) {
  long long i = (long long)blockIdx.x * blockDim.x + threadIdx.x;
  if (i >= (long long)M * HDIM) return;
  int c = (int)(i & (HDIM - 1));
  float inv = 1.0f / (float)M;
  float mu = stats[c] * inv;
  float var = stats[HDIM + c] * inv - mu * mu;
  x[i] = (x[i] - mu) * rsqrtf(var + EPSBN) * g[c] + b[c];
}

// ---------------------------------------------------------------------------
// 4) Fused classifier: z = relu(concat(hv[ev], hd[ed]) @ W1 + b1); out = sigmoid(z@W2+b2)
//    Block = 256 threads handles 16 label edges; WMMA over K=256, LDS handoff for stage 2.
// ---------------------------------------------------------------------------
__global__ void clf_kernel(const float* __restrict__ hv, const float* __restrict__ hd,
                           const int* __restrict__ eli,   // [2,EL]: [0..EL)=visit, [EL..2EL)=drug
                           const float* __restrict__ W1, const float* __restrict__ b1,
                           const float* __restrict__ W2, const float* __restrict__ b2,
                           float* __restrict__ out, int EL) {
  __shared__ float z1[16][HDIM];
  int wave = threadIdx.x >> 5;
  int row_base = blockIdx.x * 16;
  int col_base = wave * 16;
  int lane = threadIdx.x & 31;
  int hi = lane >> 4;
  int r = row_base + (lane & 15);
  if (r > EL - 1) r = EL - 1;
  const float* rowv = hv + (size_t)eli[r] * HDIM;
  const float* rowd = hd + (size_t)eli[EL + r] * HDIM;
  v8f acc = {};
#pragma unroll
  for (int k0 = 0; k0 < 2 * HDIM; k0 += 32) {
    const float* src = (k0 < HDIM) ? (rowv + k0) : (rowd + (k0 - HDIM));
    v16bf a = load_frag_a_row(src, hi);
    v16bf b = load_frag_b(W1, HDIM, k0, col_base);
    acc = __builtin_amdgcn_wmma_f32_16x16x32_bf16(false, a, false, b, (short)0, acc, false, false);
  }
  int n = col_base + (lane & 15);
  float bias = b1[n];
#pragma unroll
  for (int j = 0; j < 8; ++j) {
    float v = acc[j] + bias;
    z1[hi * 8 + j][n] = (v > 0.0f) ? v : 0.0f;
  }
  __syncthreads();
  if (threadIdx.x < 16) {
    int rr = row_base + (int)threadIdx.x;
    if (rr < EL) {
      float s = b2[0];
#pragma unroll 8
      for (int c = 0; c < HDIM; ++c) s += z1[threadIdx.x][c] * W2[c];
      out[rr] = 1.0f / (1.0f + __expf(-s));
    }
  }
}

// ---------------------------------------------------------------------------
// Host orchestration
// ---------------------------------------------------------------------------
extern "C" void kernel_launch(void* const* d_in, const int* in_sizes, int n_in,
                              void* d_out, int out_size, void* d_ws, size_t ws_size,
                              hipStream_t stream) {
  const float* x_visit = (const float*)d_in[0];
  const float* x_drug  = (const float*)d_in[1];
  const int* ei_vd = (const int*)d_in[2];   // row0 = visit src, row1 = drug dst
  const int* ei_dv = (const int*)d_in[3];   // row0 = drug src,  row1 = visit dst
  const int* eli   = (const int*)d_in[4];   // row0 = visit, row1 = drug

  const long long NV = in_sizes[0] / HDIM;
  const long long ND = in_sizes[1] / HDIM;
  const long long E  = in_sizes[2] / 2;
  const long long EL = in_sizes[4] / 2;

  // params: d_in[5 + 10*l + k], then classifier at 45..48
  const float** P = (const float**)(d_in + 5);
#define PRM(l, k) ((const float*)P[(l) * 10 + (k)])
  const float* clf_W1 = (const float*)P[40];
  const float* clf_b1 = (const float*)P[41];
  const float* clf_W2 = (const float*)P[42];
  const float* clf_b2 = (const float*)P[43];

  // workspace partition
  char* p = (char*)d_ws;
  const size_t SZ_HV = (size_t)NV * HDIM * sizeof(float);
  const size_t SZ_HD = (size_t)ND * HDIM * sizeof(float);
  float* hv_a   = (float*)p; p += SZ_HV;
  float* hv_b   = (float*)p; p += SZ_HV;
  float* mean_v = (float*)p; p += SZ_HV;
  float* hd_a   = (float*)p; p += SZ_HD;
  float* hd_b   = (float*)p; p += SZ_HD;
  float* mean_d = (float*)p; p += SZ_HD;
  float* cnt_v  = (float*)p; p += (size_t)NV * sizeof(float);
  float* cnt_d  = (float*)p; p += (size_t)ND * sizeof(float);
  float* stats  = (float*)p; p += 2 * HDIM * sizeof(float);

  const int TB = 256;
  const int grid_edge = (int)((E * 32 + TB - 1) / TB);
  const int grid_mv   = (int)(((long long)NV * HDIM + TB - 1) / TB);
  const int grid_md   = (int)(((long long)ND * HDIM + TB - 1) / TB);
  const int grid_gv   = (int)((NV + 15) / 16);
  const int grid_gd   = (int)((ND + 15) / 16);
  const int grid_bv   = (int)((NV + BN_ROWS - 1) / BN_ROWS);
  const int grid_bd   = (int)((ND + BN_ROWS - 1) / BN_ROWS);

  const float* cur_v = x_visit;
  const float* cur_d = x_drug;

  for (int l = 0; l < 4; ++l) {
    float* next_v = (l & 1) ? hv_b : hv_a;
    float* next_d = (l & 1) ? hd_b : hd_a;

    // --- scatter-mean: drugs -> visits (edge type dv) and visits -> drugs (vd)
    hipMemsetAsync(mean_v, 0, SZ_HV, stream);
    hipMemsetAsync(cnt_v, 0, (size_t)NV * sizeof(float), stream);
    hipMemsetAsync(mean_d, 0, SZ_HD, stream);
    hipMemsetAsync(cnt_d, 0, (size_t)ND * sizeof(float), stream);
    scatter_add_kernel<<<grid_edge, TB, 0, stream>>>(cur_d, ei_dv, mean_v, cnt_v, E);
    scatter_add_kernel<<<grid_edge, TB, 0, stream>>>(cur_v, ei_vd, mean_d, cnt_d, E);
    mean_div_kernel<<<grid_mv, TB, 0, stream>>>(mean_v, cnt_v, (long long)NV * HDIM);
    mean_div_kernel<<<grid_md, TB, 0, stream>>>(mean_d, cnt_d, (long long)ND * HDIM);

    // --- SAGE GEMMs + leaky_relu (WMMA bf16, fp32 accum)
    // visits updated by conv_dv (params k=3..5); drugs by conv_vd (k=0..2)
    sage_gemm_lrelu_kernel<<<grid_gv, TB, 0, stream>>>(
        mean_v, cur_v, PRM(l, 3), PRM(l, 5), PRM(l, 4), next_v, (int)NV);
    sage_gemm_lrelu_kernel<<<grid_gd, TB, 0, stream>>>(
        mean_d, cur_d, PRM(l, 0), PRM(l, 2), PRM(l, 1), next_d, (int)ND);

    // --- BatchNorm (visit: k=6,7 ; drug: k=8,9), in place
    hipMemsetAsync(stats, 0, 2 * HDIM * sizeof(float), stream);
    bn_stats_kernel<<<grid_bv, HDIM, 0, stream>>>(next_v, stats, (int)NV);
    bn_norm_kernel<<<grid_mv, TB, 0, stream>>>(next_v, stats, PRM(l, 6), PRM(l, 7), (int)NV);

    hipMemsetAsync(stats, 0, 2 * HDIM * sizeof(float), stream);
    bn_stats_kernel<<<grid_bd, HDIM, 0, stream>>>(next_d, stats, (int)ND);
    bn_norm_kernel<<<grid_md, TB, 0, stream>>>(next_d, stats, PRM(l, 8), PRM(l, 9), (int)ND);

    cur_v = next_v;
    cur_d = next_d;
  }

  // --- fused classifier head
  const int grid_clf = (int)((EL + 15) / 16);
  clf_kernel<<<grid_clf, TB, 0, stream>>>(cur_v, cur_d, eli,
                                          clf_W1, clf_b1, clf_W2, clf_b2,
                                          (float*)d_out, (int)EL);
}
